// PHASE_14809047236854
// MI455X (gfx1250) — matrix operand order repaired and checked by reading, not compiled, not run on hardware
//
#include <hip/hip_runtime.h>
#include <hip/hip_bf16.h>
#include <math.h>

typedef __bf16 bf16;
typedef __attribute__((ext_vector_type(16))) __bf16 v16bf;
typedef __attribute__((ext_vector_type(8)))  __bf16 v8bf;
typedef __attribute__((ext_vector_type(8)))  float  v8f;

#define B_    2048
#define T_    128
#define F_    128
#define H_    512
#define L_    8
#define KW_   10          // K (ring depth / conv kernel)
#define CH_   64          // CHUNK = H/L
#define GATES_ 2064       // 4H + 2L
#define NPAD  2112        // GATES padded to multiple of 64 (64-wide N tiles)
#define KPAD  672         // 642 (F+1 + H+1) padded to multiple of 32
#define HK_   5120        // H * KW

// ---------------------------------------------------------------------------
// WMMA fragment loaders (CDNA5 16x16x32 bf16 layouts, wave32)
// A (16x32, row-major in memory): lane = hi*16 + m ; element j -> K = (j&7) + 8*hi + 16*(j>>3)
//   => two contiguous 8-elem (16B) chunks at k0+8*hi and k0+16+8*hi.
// B (32x16) loaded from BT (N-major, N x K row-major): element j -> K = (j&7) + 8*hi + 16*(j>>3), N = lane&15
//   => two contiguous 8-elem (16B) chunks at BT[n][k0+8*hi] and BT[n][k0+16+8*hi].
// ---------------------------------------------------------------------------
__device__ __forceinline__ v16bf load_a_frag(const bf16* __restrict__ A, int lda,
                                             int m0, int k0, int lane) {
  const int m  = m0 + (lane & 15);
  const int hi = (lane >> 4) & 1;
  const bf16* p = A + (size_t)m * lda + k0 + 8 * hi;
  v8bf lo = *(const v8bf*)(p);
  v8bf hv = *(const v8bf*)(p + 16);
  v16bf r;
#pragma unroll
  for (int j = 0; j < 8; ++j) { r[j] = lo[j]; r[j + 8] = hv[j]; }
  return r;
}

__device__ __forceinline__ v16bf load_b_frag(const bf16* __restrict__ BT, int ldb,
                                             int n0, int k0, int lane) {
  const int n  = n0 + (lane & 15);
  const int hi = (lane >> 4) & 1;
  const bf16* p = BT + (size_t)n * ldb + k0 + 8 * hi;
  v8bf lo = *(const v8bf*)(p);
  v8bf hv = *(const v8bf*)(p + 16);
  v16bf r;
#pragma unroll
  for (int j = 0; j < 8; ++j) { r[j] = lo[j]; r[j + 8] = hv[j]; }
  return r;
}

// ---------------------------------------------------------------------------
// Generic bf16 WMMA GEMM:  C[M,N] (f32) = A[M,K] * BT[N,K]^T + bias[N]
// One wave computes a 64x64 tile (4x4 of 16x16): 16 WMMAs per 32-deep K step
// against 16 b128 loads -> 32 FLOP/byte from L2. 8 waves / block.
// M,N multiples of 64; K multiple of 32.
// ---------------------------------------------------------------------------
__global__ __launch_bounds__(256)
void gemm_bf16_wmma(const bf16* __restrict__ A, int lda,
                    const bf16* __restrict__ BT, int ldb,
                    const float* __restrict__ bias,
                    float* __restrict__ C, int ldc,
                    int tiles_m, int tiles_n, int kdim) {
  const int lane = threadIdx.x & 31;
  const int wave = threadIdx.x >> 5;
  const int wg   = blockIdx.x * 8 + wave;
  const int tm   = wg / tiles_n;
  const int tn   = wg - tm * tiles_n;
  if (tm >= tiles_m) return;               // wave-uniform guard (EXEC stays all-ones)
  const int m0 = tm * 64;
  const int n0 = tn * 64;

  v8f acc[4][4];
#pragma unroll
  for (int mi = 0; mi < 4; ++mi)
#pragma unroll
    for (int ni = 0; ni < 4; ++ni) acc[mi][ni] = {};

  for (int k0 = 0; k0 < kdim; k0 += 32) {
    v16bf fa[4];
    v16bf fb[4];
#pragma unroll
    for (int mi = 0; mi < 4; ++mi) fa[mi] = load_a_frag(A, lda, m0 + 16 * mi, k0, lane);
#pragma unroll
    for (int ni = 0; ni < 4; ++ni) fb[ni] = load_b_frag(BT, ldb, n0 + 16 * ni, k0, lane);
#pragma unroll
    for (int mi = 0; mi < 4; ++mi)
#pragma unroll
      for (int ni = 0; ni < 4; ++ni)
        acc[mi][ni] = __builtin_amdgcn_wmma_f32_16x16x32_bf16(
            false, fa[mi], false, fb[ni], (short)0, acc[mi][ni], false, false);
  }

  // C/D layout: VGPR r holds row (r + 8*hi), column = lane&15.
  const int col = lane & 15;
  const int hi  = (lane >> 4) & 1;
  float bb[4];
#pragma unroll
  for (int ni = 0; ni < 4; ++ni) bb[ni] = bias ? bias[n0 + 16 * ni + col] : 0.0f;
#pragma unroll
  for (int mi = 0; mi < 4; ++mi) {
#pragma unroll
    for (int r = 0; r < 8; ++r) {
      const size_t rowoff = (size_t)(m0 + 16 * mi + r + 8 * hi) * ldc + n0 + col;
#pragma unroll
      for (int ni = 0; ni < 4; ++ni)
        C[rowoff + 16 * ni] = acc[mi][ni][r] + bb[ni];
    }
  }
}

// ---------------------------------------------------------------------------
// Setup kernels (run once per launch)
// ---------------------------------------------------------------------------
__global__ void zero_bf16_kernel(bf16* p, int n) {
  int i = blockIdx.x * blockDim.x + threadIdx.x;
  if (i < n) p[i] = (bf16)0.0f;
}
__global__ void zero_f32_kernel(float* p, int n) {
  int i = blockIdx.x * blockDim.x + threadIdx.x;
  if (i < n) p[i] = 0.0f;
}

// Build WcatT[N=NPAD][K=KPAD] (bf16, N-major) from Wk (129 x 2064) and Wr (513 x 2064),
// and biascat[N] = bk + br.  A-column map: [0..127]=x, 128=dt, [129..640]=h, 641=dt, rest 0.
__global__ void build_wcat_kernel(const float* __restrict__ Wk, const float* __restrict__ Wr,
                                  const float* __restrict__ bk, const float* __restrict__ br,
                                  bf16* __restrict__ WcatT, float* __restrict__ biascat) {
  int i = blockIdx.x * blockDim.x + threadIdx.x;
  if (i >= NPAD * KPAD) return;
  int n = i / KPAD;
  int k = i - n * KPAD;
  float v = 0.0f;
  if (n < GATES_) {
    if (k < 129)       v = Wk[(size_t)k * GATES_ + n];
    else if (k < 641)  v = Wr[(size_t)(k - 129) * GATES_ + n];
    else if (k == 641) v = Wr[(size_t)512 * GATES_ + n];
  }
  WcatT[i] = (bf16)v;
  if (k == 0) biascat[n] = (n < GATES_) ? (bk[n] + br[n]) : 0.0f;
}

// conv_w (O=512, H=512, K=10) row-major is already N-major over hk=5120: straight convert.
__global__ void build_wconv_kernel(const float* __restrict__ conv_w, bf16* __restrict__ WconvT) {
  int i = blockIdx.x * blockDim.x + threadIdx.x;
  if (i < H_ * HK_) WconvT[i] = (bf16)conv_w[i];
}

// ---------------------------------------------------------------------------
// Per-step: pack x_t and dt columns of A (bf16). h columns come from gate kernel.
// ---------------------------------------------------------------------------
__global__ void pack_x_kernel(const float* __restrict__ x, const float* __restrict__ time,
                              bf16* __restrict__ A, int t) {
  int i = blockIdx.x * blockDim.x + threadIdx.x;
  if (i >= B_ * 130) return;
  int b = i / 130;
  int j = i - b * 130;
  if (j < 128) {
    A[(size_t)b * KPAD + j] = (bf16)x[((size_t)b * T_ + t) * F_ + j];
  } else {
    float dt = time[(size_t)b * T_ + t];
    A[(size_t)b * KPAD + (j == 128 ? 128 : 641)] = (bf16)dt;
  }
}

// ---------------------------------------------------------------------------
// Per-step gate/elementwise update. 256 threads = 4 batch rows x 64 lanes.
// ---------------------------------------------------------------------------
__device__ __forceinline__ float sigm(float v) { return 1.0f / (1.0f + expf(-v)); }

__global__ __launch_bounds__(256)
void gate_update_kernel(const float* __restrict__ xo, float* __restrict__ c,
                        float* __restrict__ hbuf, float* __restrict__ dbuf,
                        bf16* __restrict__ A, float* __restrict__ dist_out, int t) {
  const int tid = threadIdx.x;
  const int b   = blockIdx.x * 4 + (tid >> 6);
  const int ln  = tid & 63;
  const float* row = xo + (size_t)b * NPAD;

  // forget-master: cumsum(softmax(row[0:8])); input-master: reversed cumsum(softmax(row[8:16]))
  float z1[L_], z2[L_];
  float m1 = -1e30f, m2 = -1e30f;
#pragma unroll
  for (int j = 0; j < L_; ++j) {
    z1[j] = row[j];     m1 = fmaxf(m1, z1[j]);
    z2[j] = row[8 + j]; m2 = fmaxf(m2, z2[j]);
  }
  float s1 = 0.0f, s2 = 0.0f;
#pragma unroll
  for (int j = 0; j < L_; ++j) {
    z1[j] = expf(z1[j] - m1); s1 += z1[j];
    z2[j] = expf(z2[j] - m2); s2 += z2[j];
  }
  const float r1 = 1.0f / s1, r2 = 1.0f / s2;
  float fm[L_], im[L_];
  float run = 0.0f;
#pragma unroll
  for (int j = 0; j < L_; ++j) { run += z1[j]; fm[j] = run * r1; }
  run = 0.0f;
#pragma unroll
  for (int j = L_ - 1; j >= 0; --j) { run += z2[j]; im[j] = run * r2; }

  const int slot = t % KW_;
  if (ln == 0) {
    float fsum = 0.0f;
#pragma unroll
    for (int j = 0; j < L_; ++j) fsum += fm[j];
    const float d = 1.0f - 0.125f * fsum;
    dist_out[(size_t)t * B_ + b] = d;       // distance output (T,B)
    dbuf[(size_t)slot * B_ + b]  = d;       // ring for local_dis
  }

#pragma unroll
  for (int l = 0; l < L_; ++l) {
    const int ci = l * CH_ + ln;
    const float f   = sigm(row[16 + l * CH_ + ln]);
    const float ii  = sigm(row[16 + 512 + l * CH_ + ln]);
    const float oo  = sigm(row[16 + 1024 + l * CH_ + ln]);
    const float cin = tanhf(row[16 + 1536 + l * CH_ + ln]);
    const float cl  = c[(size_t)b * H_ + ci];
    const float ov  = fm[l] * im[l];
    const float cn  = ov * (f * cl + ii * cin) + (fm[l] - ov) * cl + (im[l] - ov) * cin;
    const float hn  = oo * tanhf(cn);
    c[(size_t)b * H_ + ci] = cn;
    hbuf[((size_t)slot * B_ + b) * H_ + ci] = hn;           // f32 ring for tail
    A[(size_t)b * KPAD + 129 + ci] = (bf16)hn;              // bf16 h for next-step GEMM
  }
}

// ---------------------------------------------------------------------------
// Tail: local_dis softmax + pack local_h (bf16, B x 5120) + mean over K.
// ---------------------------------------------------------------------------
__global__ __launch_bounds__(256)
void combine_kernel(const float* __restrict__ hbuf, const float* __restrict__ dbuf,
                    bf16* __restrict__ Aconv, float* __restrict__ mh) {
  const int b   = blockIdx.x;
  const int tid = threadIdx.x;
  __shared__ float ld[KW_];
  if (tid == 0) {
    float v[KW_];
    float cum = 0.0f, mx = -1e30f;
#pragma unroll
    for (int k = 0; k < KW_; ++k) {
      const int slot = (T_ - KW_ + k) % KW_;
      cum += dbuf[(size_t)slot * B_ + b];
      v[k] = cum; mx = fmaxf(mx, cum);
    }
    float s = 0.0f;
#pragma unroll
    for (int k = 0; k < KW_; ++k) { v[k] = expf(v[k] - mx); s += v[k]; }
    const float rs = 1.0f / s;
#pragma unroll
    for (int k = 0; k < KW_; ++k) ld[k] = v[k] * rs;
  }
  __syncthreads();
  for (int h = tid; h < H_; h += 256) {
    float sum = 0.0f;
#pragma unroll
    for (int k = 0; k < KW_; ++k) {
      const int slot = (T_ - KW_ + k) % KW_;
      const float lh = hbuf[((size_t)slot * B_ + b) * H_ + h] * ld[k];
      Aconv[(size_t)b * HK_ + h * KW_ + k] = (bf16)lh;
      sum += lh;
    }
    mh[(size_t)b * H_ + h] = sum * (1.0f / KW_);
  }
}

// ---------------------------------------------------------------------------
// Tail: theme MLP (512 -> 85 -> 512) + final multiply with conv_out.
// ---------------------------------------------------------------------------
__global__ __launch_bounds__(512)
void theme_out_kernel(const float* __restrict__ mh, const float* __restrict__ convout,
                      const float* __restrict__ Ws, const float* __restrict__ bs,
                      const float* __restrict__ Wrs, const float* __restrict__ brs,
                      float* __restrict__ out) {
  const int b   = blockIdx.x;
  const int tid = threadIdx.x;
  __shared__ float smh[H_];
  __shared__ float st[96];
  smh[tid] = mh[(size_t)b * H_ + tid];
  __syncthreads();
  if (tid < 85) {
    float a = bs[tid];
    for (int hh = 0; hh < H_; ++hh) a += smh[hh] * Ws[(size_t)hh * 85 + tid];
    st[tid] = fmaxf(a, 0.0f);
  }
  __syncthreads();
  float a = brs[tid];
#pragma unroll 5
  for (int j = 0; j < 85; ++j) a += st[j] * Wrs[(size_t)j * H_ + tid];
  const float theme = 1.0f / (1.0f + expf(-a));
  out[(size_t)b * H_ + tid] = theme * convout[(size_t)b * H_ + tid];
}

// ---------------------------------------------------------------------------
// Host launcher
// ---------------------------------------------------------------------------
extern "C" void kernel_launch(void* const* d_in, const int* in_sizes, int n_in,
                              void* d_out, int out_size, void* d_ws, size_t ws_size,
                              hipStream_t stream) {
  (void)in_sizes; (void)n_in; (void)out_size; (void)ws_size;
  const float* x      = (const float*)d_in[0];
  const float* time_  = (const float*)d_in[1];
  const float* Wk     = (const float*)d_in[2];
  const float* bk     = (const float*)d_in[3];
  const float* Wr     = (const float*)d_in[4];
  const float* br     = (const float*)d_in[5];
  const float* Ws     = (const float*)d_in[6];
  const float* bs     = (const float*)d_in[7];
  const float* Wrs    = (const float*)d_in[8];
  const float* brs    = (const float*)d_in[9];
  const float* conv_w = (const float*)d_in[10];
  const float* conv_b = (const float*)d_in[11];

  float* out_main = (float*)d_out;                      // (B, H)
  float* out_dist = (float*)d_out + (size_t)B_ * H_;    // (T, B)

  // Workspace carve-up (256B aligned slices)
  size_t off = 0;
  auto carve = [&](size_t bytes) -> void* {
    void* p = (char*)d_ws + off;
    off += (bytes + 255) & ~(size_t)255;
    return p;
  };
  bf16*  A       = (bf16*) carve((size_t)B_ * KPAD * sizeof(bf16));
  bf16*  WcatT   = (bf16*) carve((size_t)NPAD * KPAD * sizeof(bf16));
  bf16*  WconvT  = (bf16*) carve((size_t)H_ * HK_ * sizeof(bf16));
  float* biascat = (float*)carve((size_t)NPAD * sizeof(float));
  float* xo      = (float*)carve((size_t)B_ * NPAD * sizeof(float));
  float* cbuf    = (float*)carve((size_t)B_ * H_ * sizeof(float));
  float* hbuf    = (float*)carve((size_t)KW_ * B_ * H_ * sizeof(float));
  float* dbuf    = (float*)carve((size_t)KW_ * B_ * sizeof(float));
  bf16*  Aconv   = (bf16*) carve((size_t)B_ * HK_ * sizeof(bf16));
  float* mh      = (float*)carve((size_t)B_ * H_ * sizeof(float));
  float* convout = (float*)carve((size_t)B_ * H_ * sizeof(float));

  // ---- once-per-launch setup (deterministic; rebuilt every call) ----
  {
    int nA = B_ * KPAD;
    zero_bf16_kernel<<<(nA + 255) / 256, 256, 0, stream>>>(A, nA);
    int nC = B_ * H_;
    zero_f32_kernel<<<(nC + 255) / 256, 256, 0, stream>>>(cbuf, nC);
    int nW = NPAD * KPAD;
    build_wcat_kernel<<<(nW + 255) / 256, 256, 0, stream>>>(Wk, Wr, bk, br, WcatT, biascat);
    int nV = H_ * HK_;
    build_wconv_kernel<<<(nV + 255) / 256, 256, 0, stream>>>(conv_w, WconvT);
  }

  // ---- recurrent loop: pack -> WMMA GEMM -> gate update ----
  const int tiles_m = B_ / 64;          // 32
  const int tiles_n = NPAD / 64;        // 33
  const int gemm_blocks = (tiles_m * tiles_n + 7) / 8;   // 132
  for (int t = 0; t < T_; ++t) {
    int nP = B_ * 130;
    pack_x_kernel<<<(nP + 255) / 256, 256, 0, stream>>>(x, time_, A, t);
    gemm_bf16_wmma<<<gemm_blocks, 256, 0, stream>>>(A, KPAD, WcatT, KPAD, biascat,
                                                    xo, NPAD, tiles_m, tiles_n, KPAD);
    gate_update_kernel<<<B_ / 4, 256, 0, stream>>>(xo, cbuf, hbuf, dbuf, A, out_dist, t);
  }

  // ---- tail: local weighting, conv einsum (WMMA), theme MLP ----
  combine_kernel<<<B_, 256, 0, stream>>>(hbuf, dbuf, Aconv, mh);
  {
    const int tm = B_ / 64;             // 32
    const int tn = H_ / 64;             // 8
    gemm_bf16_wmma<<<(tm * tn + 7) / 8, 256, 0, stream>>>(Aconv, HK_, WconvT, HK_, conv_b,
                                                          convout, H_, tm, tn, HK_);
  }
  theme_out_kernel<<<B_, 512, 0, stream>>>(mh, convout, Ws, bs, Wrs, brs, out_main);
}
